// ApaBlock_69904887710395
// MI455X (gfx1250) — matrix operand order repaired and checked by reading, not compiled, not run on hardware
//
#include <hip/hip_runtime.h>

// Problem constants (B, IN, H, OUT, RANK) = (2048, 256, 128, 128, 8)
#define EPS_BN 1e-5f

typedef __attribute__((ext_vector_type(16))) __bf16 bf16x16;
typedef __attribute__((ext_vector_type(8)))  float  f32x8;
typedef __attribute__((ext_vector_type(4)))  float  f32x4;
typedef __attribute__((ext_vector_type(4)))  int    i32x4;

#define WMMA_BF16(a, b, c) \
  __builtin_amdgcn_wmma_f32_16x16x32_bf16(false, (a), false, (b), (short)0, (c), false, false)

// ---- async global->LDS staging (CDNA5 ASYNCcnt path), guarded ---------------
#if __has_builtin(__builtin_amdgcn_global_load_async_to_lds_b128) && \
    __has_builtin(__builtin_amdgcn_s_wait_asynccnt)
#define USE_ASYNC_LDS 1
#else
#define USE_ASYNC_LDS 0
#endif

typedef __attribute__((address_space(1))) i32x4 gv4i;  // global (AS1 / __device__)
typedef __attribute__((address_space(3))) i32x4 lv4i;  // LDS    (AS3 / __shared__)

// Copy one contiguous 32KB slab (2048 x 16B) with 256 threads.
__device__ __forceinline__ void stage_32k(__bf16* dst, const __bf16* src, int tid) {
#if USE_ASYNC_LDS
  gv4i* g = (gv4i*)src;
  lv4i* l = (lv4i*)dst;
#pragma unroll
  for (int j = 0; j < 8; ++j) {  // 8 x b128 per thread -> ASYNCcnt += 8 per wave
    int idx = tid + j * 256;
    __builtin_amdgcn_global_load_async_to_lds_b128(g + idx, l + idx, 0, 0);
  }
#else
  for (int i = tid; i < 2048; i += 256)
    ((f32x4*)dst)[i] = ((const f32x4*)src)[i];
#endif
}
__device__ __forceinline__ void stage_wait_8() {
#if USE_ASYNC_LDS
  __builtin_amdgcn_s_wait_asynccnt(8);
#endif
}
__device__ __forceinline__ void stage_wait_0() {
#if USE_ASYNC_LDS
  __builtin_amdgcn_s_wait_asynccnt(0);
#endif
}

// ---- WMMA fragment loaders (CDNA5 16-bit layouts, wave32) -------------------
// A 16x32: lane holds row (lane&15); K elems 0..7 & 16..23 (lanes<16),
//          8..15 & 24..31 (lanes>=16). Two 16B loads at +0 and +16 elements.
__device__ __forceinline__ bf16x16 load_a_frag(const __bf16* s, int row0, int ld,
                                               int k0, int lane) {
  const __bf16* p = s + (size_t)(row0 + (lane & 15)) * ld + k0 + ((lane & 16) ? 8 : 0);
  bf16x16 r;
  ((f32x4*)&r)[0] = *(const f32x4*)p;
  ((f32x4*)&r)[1] = *(const f32x4*)(p + 16);
  return r;
}
// B 32x16: lane holds column (lane&15); K elems 0..7 & 16..23 (lanes<16),
//          8..15 & 24..31 (lanes>=16). Two 16B loads at +0 and +16 elements.
__device__ __forceinline__ bf16x16 load_b_frag(const __bf16* s, int col0, int ld,
                                               int k0, int lane) {
  const __bf16* p = s + (size_t)(col0 + (lane & 15)) * ld + k0 + ((lane & 16) ? 8 : 0);
  bf16x16 r;
  ((f32x4*)&r)[0] = *(const f32x4*)p;
  ((f32x4*)&r)[1] = *(const f32x4*)(p + 16);
  return r;
}
__device__ __forceinline__ bf16x16 scale_frag(bf16x16 v, float s) {
  bf16x16 r;
#pragma unroll
  for (int i = 0; i < 16; ++i) r[i] = (__bf16)((float)v[i] * s);
  return r;
}

// ---- prep: P[i][p][q][k] -> Pt2[i][p][k][q] in bf16 -------------------------
__global__ __launch_bounds__(256) void k_prep_p(const float* __restrict__ P,
                                                __bf16* __restrict__ Pt2) {
  size_t base = (size_t)blockIdx.x * 16384;  // one (i,p) slab of 128x128
  for (int e = threadIdx.x; e < 16384; e += 256) {
    int q = e >> 7, k = e & 127;
    Pt2[base + (size_t)k * 128 + q] = (__bf16)P[base + e];
  }
}

// ---- prep: X->bf16; W1,W2 (256x128) and W3 (128x128) -> transposed bf16 -----
__global__ __launch_bounds__(256) void k_prep_misc(
    const float* __restrict__ X,  const float* __restrict__ W1,
    const float* __restrict__ W2, const float* __restrict__ W3,
    __bf16* __restrict__ Xb, __bf16* __restrict__ W1t,
    __bf16* __restrict__ W2t, __bf16* __restrict__ W3t) {
  int i = blockIdx.x * 256 + threadIdx.x;
  if (i < 524288) {
    Xb[i] = (__bf16)X[i];
  } else if (i < 524288 + 32768) {
    int e = i - 524288; int k = e >> 7, n = e & 127;
    W1t[(size_t)n * 256 + k] = (__bf16)W1[e];
  } else if (i < 524288 + 65536) {
    int e = i - 524288 - 32768; int k = e >> 7, n = e & 127;
    W2t[(size_t)n * 256 + k] = (__bf16)W2[e];
  } else if (i < 524288 + 81920) {
    int e = i - 524288 - 65536; int k = e >> 7, n = e & 127;
    W3t[(size_t)n * 128 + k] = (__bf16)W3[e];
  }
}

// ---- Z = relu(X @ W1 + b1): M=2048, N=128, K=256 ---------------------------
// 256 threads = 8 waves; wave (wr,wc) covers rows wr*32..+32, cols wc*64..+64.
__global__ __launch_bounds__(256) void k_gemm_z(
    const __bf16* __restrict__ Xb, const __bf16* __restrict__ W1t,
    const float* __restrict__ b1, float* __restrict__ Zi, __bf16* __restrict__ Zb) {
  extern __shared__ char smem[];
  __bf16* A_s = (__bf16*)smem;            // 128 x 128 bf16
  __bf16* B_s = (__bf16*)(smem + 32768);  // 128(n) x 128(k) bf16
  const int tid = threadIdx.x, lane = tid & 31, wave = tid >> 5;
  const int wr = wave >> 1, wc = wave & 1;
  const int b0 = blockIdx.x * 128;
  f32x8 acc[2][4] = {};
  for (int c = 0; c < 2; ++c) {  // K chunks of 128
    for (int i = tid; i < 2048; i += 256) {
      int r = i >> 4, v = i & 15;
      ((f32x4*)(A_s + r * 128))[v] =
          ((const f32x4*)(Xb + (size_t)(b0 + r) * 256 + c * 128))[v];
      ((f32x4*)(B_s + r * 128))[v] =
          ((const f32x4*)(W1t + (size_t)r * 256 + c * 128))[v];
    }
    __syncthreads();
#pragma unroll
    for (int ks = 0; ks < 4; ++ks) {
      int k0 = ks * 32;
      bf16x16 a0 = load_a_frag(A_s, wr * 32,      128, k0, lane);
      bf16x16 a1 = load_a_frag(A_s, wr * 32 + 16, 128, k0, lane);
#pragma unroll
      for (int t = 0; t < 4; ++t) {
        bf16x16 bf = load_b_frag(B_s, wc * 64 + t * 16, 128, k0, lane);
        acc[0][t] = WMMA_BF16(a0, bf, acc[0][t]);
        acc[1][t] = WMMA_BF16(a1, bf, acc[1][t]);
      }
    }
    __syncthreads();
  }
#pragma unroll
  for (int tr = 0; tr < 2; ++tr)
#pragma unroll
    for (int tc = 0; tc < 4; ++tc) {
      int col = wc * 64 + tc * 16 + (lane & 15);
      float bias = b1[col];
      int rbase = b0 + wr * 32 + tr * 16 + ((lane & 16) ? 8 : 0);
#pragma unroll
      for (int v = 0; v < 8; ++v) {
        float val = fmaxf(acc[tr][tc][v] + bias, 0.0f);
        size_t idx = (size_t)(rbase + v) * 128 + col;
        Zi[idx] = val;           // scan carry (f32)
        Zb[idx] = (__bf16)val;   // fixed Z operand (bf16) for the quadratic
      }
    }
}

// ---- quadratic GEMM, one rank: Zt_part[s] = (Zi ox Z) @ P_i  ---------------
// Per block: M-tile = 128 rows, all N=128 cols, p in [p0, p0+16) (split-K=8).
// A fragment built in registers: a[b][q] = Zi[b,p] * Z[b,q].
// P slabs are double-buffered in LDS via async global->LDS copies (ASYNCcnt).
__global__ __launch_bounds__(256) void k_gemm_quad(
    const float* __restrict__ Zi, const __bf16* __restrict__ Zb,
    const __bf16* __restrict__ Pt2i,  // this rank: [p][k][q] bf16
    float* __restrict__ Zt_part) {
  extern __shared__ char smem[];
  __bf16* Zb_s   = (__bf16*)smem;            // 128 x 128 bf16 (Z tile, loaded once)
  __bf16* Bbuf0  = (__bf16*)(smem + 32768);  // P slab buffer 0: 128(n) x 128(q)
  __bf16* Bbuf1  = (__bf16*)(smem + 65536);  // P slab buffer 1
  const int tid = threadIdx.x, lane = tid & 31, wave = tid >> 5;
  const int wr = wave >> 1, wc = wave & 1;
  const int b0 = blockIdx.x * 128;
  const int p0 = blockIdx.y * 16;

  // prologue: Z tile + first P slab in flight
  stage_32k(Zb_s, Zb + (size_t)b0 * 128, tid);
  stage_32k(Bbuf0, Pt2i + (size_t)p0 * 16384, tid);

  f32x8 acc[2][4] = {};
  for (int pp = 0; pp < 16; ++pp) {
    __bf16* B_s = (pp & 1) ? Bbuf1 : Bbuf0;
    __bf16* B_n = (pp & 1) ? Bbuf0 : Bbuf1;
    int p = p0 + pp;
    if (pp + 1 < 16) {
      stage_32k(B_n, Pt2i + (size_t)(p + 1) * 16384, tid);  // next slab in flight
      __builtin_prefetch((const void*)(Pt2i + (size_t)(p + 2) * 16384 + tid * 64), 0, 1);
      stage_wait_8();  // drain everything older than the slab just issued
    } else {
      stage_wait_0();
    }
    __syncthreads();
    float zi0 = Zi[(size_t)(b0 + wr * 32 +      (lane & 15)) * 128 + p];
    float zi1 = Zi[(size_t)(b0 + wr * 32 + 16 + (lane & 15)) * 128 + p];
#pragma unroll
    for (int ks = 0; ks < 4; ++ks) {
      int k0 = ks * 32;
      bf16x16 a0 = scale_frag(load_a_frag(Zb_s, wr * 32,      128, k0, lane), zi0);
      bf16x16 a1 = scale_frag(load_a_frag(Zb_s, wr * 32 + 16, 128, k0, lane), zi1);
#pragma unroll
      for (int t = 0; t < 4; ++t) {
        bf16x16 bf = load_b_frag(B_s, wc * 64 + t * 16, 128, k0, lane);
        acc[0][t] = WMMA_BF16(a0, bf, acc[0][t]);
        acc[1][t] = WMMA_BF16(a1, bf, acc[1][t]);
      }
    }
    __syncthreads();  // before the buffer we just read gets overwritten
  }
  float* outp = Zt_part + (size_t)blockIdx.y * (2048 * 128);
#pragma unroll
  for (int tr = 0; tr < 2; ++tr)
#pragma unroll
    for (int tc = 0; tc < 4; ++tc) {
      int col = wc * 64 + tc * 16 + (lane & 15);
      int rbase = b0 + wr * 32 + tr * 16 + ((lane & 16) ? 8 : 0);
#pragma unroll
      for (int v = 0; v < 8; ++v)
        outp[(size_t)(rbase + v) * 128 + col] = acc[tr][tc][v];
    }
}

// ---- sum split-K partials ---------------------------------------------------
__global__ __launch_bounds__(256) void k_sum_parts(const float* __restrict__ parts,
                                                   float* __restrict__ Zt) {
  int i = blockIdx.x * 256 + threadIdx.x;  // 2048*128 total
  float s = 0.f;
#pragma unroll
  for (int p = 0; p < 8; ++p) s += parts[(size_t)p * 262144 + i];
  Zt[i] = s;
}

// ---- per-column batch mean/var (biased) ------------------------------------
__global__ __launch_bounds__(256) void k_colstats(const float* __restrict__ M,
                                                  float* __restrict__ stats, int rows) {
  __shared__ float s1[256], s2[256];
  int j = blockIdx.x, tid = threadIdx.x;
  float a = 0.f, b = 0.f;
  for (int r = tid; r < rows; r += 256) {
    float v = M[(size_t)r * 128 + j];
    a += v; b += v * v;
  }
  s1[tid] = a; s2[tid] = b;
  __syncthreads();
  for (int s = 128; s > 0; s >>= 1) {
    if (tid < s) { s1[tid] += s1[tid + s]; s2[tid] += s2[tid + s]; }
    __syncthreads();
  }
  if (tid == 0) {
    float m = s1[0] / rows;
    stats[j] = m;
    stats[128 + j] = s2[0] / rows - m * m;
  }
}

// ---- BN apply for scan step: Zi = bn(Zt); Y (+)= Zi/RANK -------------------
__global__ __launch_bounds__(256) void k_bn_quad(
    const float* __restrict__ Zt, const float* __restrict__ stats,
    const float* __restrict__ gz, const float* __restrict__ bz,
    float* __restrict__ Zi, float* __restrict__ Y, int first) {
  int i = blockIdx.x * 256 + threadIdx.x;
  int j = i & 127;
  float zn = (Zt[i] - stats[j]) * rsqrtf(stats[128 + j] + EPS_BN) * gz[j] + bz[j];
  Zi[i] = zn;
  float y = zn * 0.125f;
  if (first) Y[i] = y; else Y[i] += y;
}

// ---- BN apply for Y -> bf16 -------------------------------------------------
__global__ __launch_bounds__(256) void k_bn_y(
    const float* __restrict__ Y, const float* __restrict__ stats,
    const float* __restrict__ gy, const float* __restrict__ by,
    __bf16* __restrict__ Yb) {
  int i = blockIdx.x * 256 + threadIdx.x;
  int j = i & 127;
  float yn = (Y[i] - stats[j]) * rsqrtf(stats[128 + j] + EPS_BN) * gy[j] + by[j];
  Yb[i] = (__bf16)yn;
}

// ---- out = relu(relu(Y@W3+b3) + relu(X@W2+b2)) ------------------------------
__global__ __launch_bounds__(256) void k_final(
    const __bf16* __restrict__ Yb, const __bf16* __restrict__ Xb,
    const __bf16* __restrict__ W3t, const __bf16* __restrict__ W2t,
    const float* __restrict__ b3, const float* __restrict__ b2,
    float* __restrict__ out) {
  extern __shared__ char smem[];
  __bf16* A_s = (__bf16*)smem;
  __bf16* B_s = (__bf16*)(smem + 32768);
  const int tid = threadIdx.x, lane = tid & 31, wave = tid >> 5;
  const int wr = wave >> 1, wc = wave & 1;
  const int b0 = blockIdx.x * 128;
  f32x8 acc1[2][4] = {}, acc2[2][4] = {};
  // X @ W2 : K = 256 in two chunks (row-strided -> synchronous staging)
  for (int c = 0; c < 2; ++c) {
    for (int i = tid; i < 2048; i += 256) {
      int r = i >> 4, v = i & 15;
      ((f32x4*)(A_s + r * 128))[v] =
          ((const f32x4*)(Xb + (size_t)(b0 + r) * 256 + c * 128))[v];
      ((f32x4*)(B_s + r * 128))[v] =
          ((const f32x4*)(W2t + (size_t)r * 256 + c * 128))[v];
    }
    __syncthreads();
#pragma unroll
    for (int ks = 0; ks < 4; ++ks) {
      int k0 = ks * 32;
      bf16x16 a0 = load_a_frag(A_s, wr * 32,      128, k0, lane);
      bf16x16 a1 = load_a_frag(A_s, wr * 32 + 16, 128, k0, lane);
#pragma unroll
      for (int t = 0; t < 4; ++t) {
        bf16x16 bf = load_b_frag(B_s, wc * 64 + t * 16, 128, k0, lane);
        acc2[0][t] = WMMA_BF16(a0, bf, acc2[0][t]);
        acc2[1][t] = WMMA_BF16(a1, bf, acc2[1][t]);
      }
    }
    __syncthreads();
  }
  // Y @ W3 : K = 128 (flat tiles -> async staging)
  stage_32k(A_s, Yb + (size_t)b0 * 128, tid);
  stage_32k(B_s, W3t, tid);
  stage_wait_0();
  __syncthreads();
#pragma unroll
  for (int ks = 0; ks < 4; ++ks) {
    int k0 = ks * 32;
    bf16x16 a0 = load_a_frag(A_s, wr * 32,      128, k0, lane);
    bf16x16 a1 = load_a_frag(A_s, wr * 32 + 16, 128, k0, lane);
#pragma unroll
    for (int t = 0; t < 4; ++t) {
      bf16x16 bf = load_b_frag(B_s, wc * 64 + t * 16, 128, k0, lane);
      acc1[0][t] = WMMA_BF16(a0, bf, acc1[0][t]);
      acc1[1][t] = WMMA_BF16(a1, bf, acc1[1][t]);
    }
  }
#pragma unroll
  for (int tr = 0; tr < 2; ++tr)
#pragma unroll
    for (int tc = 0; tc < 4; ++tc) {
      int col = wc * 64 + tc * 16 + (lane & 15);
      float bias3 = b3[col], bias2 = b2[col];
      int rbase = b0 + wr * 32 + tr * 16 + ((lane & 16) ? 8 : 0);
#pragma unroll
      for (int v = 0; v < 8; ++v) {
        float r1 = fmaxf(acc1[tr][tc][v] + bias3, 0.0f);
        float r2 = fmaxf(acc2[tr][tc][v] + bias2, 0.0f);
        out[(size_t)(rbase + v) * 128 + col] = fmaxf(r1 + r2, 0.0f);
      }
    }
}

// ---- workspace layout (bytes) ----------------------------------------------
static const size_t OFF_PT2  = 0;                       // 8*128*128*128*2 = 33,554,432
static const size_t OFF_XB   = OFF_PT2  + 33554432;     // 1,048,576
static const size_t OFF_W1T  = OFF_XB   + 1048576;      // 65,536
static const size_t OFF_W2T  = OFF_W1T  + 65536;        // 65,536
static const size_t OFF_W3T  = OFF_W2T  + 65536;        // 32,768
static const size_t OFF_ZB   = OFF_W3T  + 32768;        // 524,288
static const size_t OFF_ZI   = OFF_ZB   + 524288;       // 1,048,576
static const size_t OFF_ZT   = OFF_ZI   + 1048576;      // 1,048,576
static const size_t OFF_Y    = OFF_ZT   + 1048576;      // 1,048,576
static const size_t OFF_YB   = OFF_Y    + 1048576;      // 524,288
static const size_t OFF_ST   = OFF_YB   + 524288;       // 1,024
static const size_t OFF_PART = OFF_ST   + 1024;         // 8*2048*128*4 = 8,388,608

extern "C" void kernel_launch(void* const* d_in, const int* in_sizes, int n_in,
                              void* d_out, int out_size, void* d_ws, size_t ws_size,
                              hipStream_t stream) {
  const float* X  = (const float*)d_in[0];
  const float* W1 = (const float*)d_in[1];
  const float* b1 = (const float*)d_in[2];
  const float* W2 = (const float*)d_in[3];
  const float* b2 = (const float*)d_in[4];
  const float* W3 = (const float*)d_in[5];
  const float* b3 = (const float*)d_in[6];
  const float* P  = (const float*)d_in[7];
  const float* gz = (const float*)d_in[8];
  const float* bz = (const float*)d_in[9];
  const float* gy = (const float*)d_in[10];
  const float* by = (const float*)d_in[11];

  char* ws = (char*)d_ws;
  __bf16* Pt2 = (__bf16*)(ws + OFF_PT2);
  __bf16* Xb  = (__bf16*)(ws + OFF_XB);
  __bf16* W1t = (__bf16*)(ws + OFF_W1T);
  __bf16* W2t = (__bf16*)(ws + OFF_W2T);
  __bf16* W3t = (__bf16*)(ws + OFF_W3T);
  __bf16* Zb  = (__bf16*)(ws + OFF_ZB);
  float*  Zi  = (float*)(ws + OFF_ZI);
  float*  Zt  = (float*)(ws + OFF_ZT);
  float*  Y   = (float*)(ws + OFF_Y);
  __bf16* Yb  = (__bf16*)(ws + OFF_YB);
  float*  st  = (float*)(ws + OFF_ST);
  float*  part = (float*)(ws + OFF_PART);

  // one-time per call conversions (deterministic; inputs never mutated)
  k_prep_p<<<1024, 256, 0, stream>>>(P, Pt2);
  k_prep_misc<<<2368, 256, 0, stream>>>(X, W1, W2, W3, Xb, W1t, W2t, W3t);

  // Z = relu(X@W1 + b1)
  k_gemm_z<<<16, 256, 65536, stream>>>(Xb, W1t, b1, Zi, Zb);

  // scan over 8 ranks
  for (int r = 0; r < 8; ++r) {
    const __bf16* Pt2i = Pt2 + (size_t)r * 128 * 16384;
    k_gemm_quad<<<dim3(16, 8), 256, 98304, stream>>>(Zi, Zb, Pt2i, part);
    k_sum_parts<<<1024, 256, 0, stream>>>(part, Zt);
    k_colstats<<<128, 256, 0, stream>>>(Zt, st, 2048);
    k_bn_quad<<<1024, 256, 0, stream>>>(Zt, st, gz, bz, Zi, Y, r == 0);
  }

  // Y = bn(Y); out = relu(relu(Y@W3+b3) + relu(X@W2+b2))
  k_colstats<<<128, 256, 0, stream>>>(Y, st, 2048);
  k_bn_y<<<1024, 256, 0, stream>>>(Y, st, gy, by, Yb);
  k_final<<<16, 256, 65536, stream>>>(Yb, Xb, W3t, W2t, b3, b2, (float*)d_out);
}
